// HomoAttModel_36550171689026
// MI455X (gfx1250) — compile-verified, matrix operand order checked by reading, not run on hardware
//
#include <hip/hip_runtime.h>
#include <stdint.h>

#define IN_DIM 256
#define OUT_DIM 64
#define NHEAD 4
#define NCOLS 256      // NHEAD * OUT_DIM
#define KDIM 256
#define DEG 32
#define NEDGE 33       // DEG + self loop
#define N2 16000
#define N1 4000
#define N0 1000
#define LALPHA 0.2f

typedef __attribute__((ext_vector_type(16))) __bf16 v16bf;
typedef __attribute__((ext_vector_type(8)))  float  v8f;

union FragAB { v16bf v; uint4 q[2]; };

__device__ __forceinline__ unsigned short f32_to_bf16(float f) {
    union { float f; uint32_t u; } c; c.f = f;
    uint32_t u = c.u;
    return (unsigned short)((u + 0x7FFFu + ((u >> 16) & 1u)) >> 16);
}

// ---------------- f32 -> bf16 convert ----------------
__global__ void cvt_f32_bf16(const float* __restrict__ in,
                             unsigned short* __restrict__ out, int n) {
    int i = blockIdx.x * blockDim.x + threadIdx.x;
    if (i < n) out[i] = f32_to_bf16(in[i]);
}

// ---- pack W[H, 256, 64] -> Bp[kt][c][kk] bf16, B[d,c] = W[h,d,o], c=h*64+o ----
__global__ void pack_W_headed(const float* __restrict__ W,
                              unsigned short* __restrict__ Bp) {
    int idx = blockIdx.x * blockDim.x + threadIdx.x;   // 8*256*32 = 65536
    int kk = idx & 31;
    int c  = (idx >> 5) & 255;
    int kt = idx >> 13;
    int d  = kt * 32 + kk;
    int h  = c >> 6, o = c & 63;
    Bp[idx] = f32_to_bf16(W[(size_t)h * IN_DIM * OUT_DIM + (size_t)d * OUT_DIM + o]);
}

// ---- pack lin_W[256,256] transposed: B[d,c] = lin_W[c,d] ----
__global__ void pack_W_linT(const float* __restrict__ W,
                            unsigned short* __restrict__ Bp) {
    int idx = blockIdx.x * blockDim.x + threadIdx.x;
    int kk = idx & 31;
    int c  = (idx >> 5) & 255;
    int kt = idx >> 13;
    int d  = kt * 32 + kk;
    Bp[idx] = f32_to_bf16(W[(size_t)c * KDIM + d]);
}

// ---------------- WMMA GEMM: C[N,256] = A[N,256] x B[256,256] ----------------
// block = 256 threads = 8 waves; wave(w): rows r0=bx*64+(w&3)*16, cols c0=by*32+(w>>2)*16
__global__ __launch_bounds__(256)
void wmma_gemm_bf16(const unsigned short* __restrict__ A,
                    const unsigned short* __restrict__ Bp,
                    const float* __restrict__ bias,
                    float* __restrict__ C, int N, int mode) {
    const int tid  = threadIdx.x;
    const int lane = tid & 31;
    const int wave = tid >> 5;
    const int half = lane >> 4;
    const int l16  = lane & 15;
    const int r0 = blockIdx.x * 64 + (wave & 3) * 16;
    const int c0 = blockIdx.y * 32 + (wave >> 2) * 16;

    int rowA = r0 + l16; if (rowA >= N) rowA = N - 1;   // clamp; masked at store
    const int colB = c0 + l16;

    const uint4* A4 = reinterpret_cast<const uint4*>(A);   // 1 uint4 = 8 bf16
    const uint4* B4 = reinterpret_cast<const uint4*>(Bp);

    v8f acc = {};
#pragma unroll
    for (int kt = 0; kt < KDIM / 32; ++kt) {
        FragAB fa, fb;
        // A fragment: lane holds row M=rowA, K = kt*32 + half*8 + [0..7] and +16
        size_t abase = ((size_t)rowA * KDIM + kt * 32 + half * 8) >> 3;
        fa.q[0] = A4[abase];
        fa.q[1] = A4[abase + 2];
        // B fragment: lane holds col colB, same K pattern; Bp[kt][col][kk]
        size_t bbase = (((size_t)kt * NCOLS + colB) * 32 + half * 8) >> 3;
        fb.q[0] = B4[bbase];
        fb.q[1] = B4[bbase + 2];
        acc = __builtin_amdgcn_wmma_f32_16x16x32_bf16(false, fa.v, false, fb.v,
                                                      (short)0, acc, false, false);
    }
#pragma unroll
    for (int i = 0; i < 8; ++i) {
        int row = r0 + half * 8 + i;
        int col = c0 + l16;
        if (row < N) {
            float v = acc[i];
            if (mode == 1) v = tanhf(v + bias[col]);
            C[(size_t)row * NCOLS + col] = v;
        }
    }
}

// ---- per-node attention scalars: st[h,n]=h_n·a[h,:64], sn[h,n]=h_n·a[h,64:] ----
__global__ __launch_bounds__(256)
void node_scores(const float* __restrict__ hcat, const float* __restrict__ a,
                 float* __restrict__ st, float* __restrict__ sn, int N) {
    __shared__ float sa[NHEAD * 2 * OUT_DIM];   // 512 floats
    int tid = threadIdx.x;
    for (int i = tid; i < NHEAD * 2 * OUT_DIM; i += blockDim.x) sa[i] = a[i];
    __syncthreads();
    int n = blockIdx.x * blockDim.x + tid;
    if (n >= N) return;
    const float* hrow = hcat + (size_t)n * NCOLS;
#pragma unroll
    for (int h = 0; h < NHEAD; ++h) {
        float s1 = 0.f, s2 = 0.f;
        for (int o = 0; o < OUT_DIM; ++o) {
            float v = hrow[h * OUT_DIM + o];
            s1 += v * sa[h * 128 + o];
            s2 += v * sa[h * 128 + 64 + o];
        }
        st[h * N + n] = s1;
        sn[h * N + n] = s2;
    }
}

// ---- per-target: dedup neighbors, leaky+softmax, weighted gather, ELU ----
__global__ __launch_bounds__(256)
void aggregate(const float* __restrict__ hcat,
               const float* __restrict__ st, const float* __restrict__ sn,
               const int* __restrict__ adj_nbr,
               float* __restrict__ xout, int T, int Nin) {
    __shared__ int   s_nbr[NEDGE];
    __shared__ int   s_valid[NEDGE];
    __shared__ float s_w[NHEAD][NEDGE];
    const int t = blockIdx.x;
    const int tid = threadIdx.x;

    if (tid < NEDGE) {
        int idx = (tid < DEG) ? (t * DEG + tid) : (T * DEG + t);   // self-loop tail
        s_nbr[tid] = adj_nbr[idx];
    }
    __syncthreads();
    if (tid < NEDGE) {                 // dense-scatter dedup (dup edges have equal e)
        int me = s_nbr[tid], v = 1;
        for (int j = 0; j < tid; ++j) if (s_nbr[j] == me) { v = 0; break; }
        s_valid[tid] = v;
    }
    __syncthreads();
    if (tid < NHEAD) {                 // softmax per head over unique neighbors
        const int h = tid;
        const float stv = st[h * Nin + t];
        float m = -3.0e38f;
        for (int i = 0; i < NEDGE; ++i) {
            if (s_valid[i]) {
                float z = stv + sn[h * Nin + s_nbr[i]];
                z = (z >= 0.f) ? z : LALPHA * z;
                m = fmaxf(m, z);
            }
        }
        float sum = 0.f;
        for (int i = 0; i < NEDGE; ++i) {
            float w = 0.f;
            if (s_valid[i]) {
                float z = stv + sn[h * Nin + s_nbr[i]];
                z = (z >= 0.f) ? z : LALPHA * z;
                w = __expf(z - m);
            }
            s_w[h][i] = w;
            sum += w;
        }
        float inv = 1.f / sum;
        for (int i = 0; i < NEDGE; ++i) s_w[h][i] *= inv;
    }
    __syncthreads();
    const int c = tid;                 // output column 0..255
    const int h = c >> 6;
    float acc = 0.f;
    for (int i = 0; i < NEDGE; ++i) {
        float w = s_w[h][i];
        if (w != 0.f) acc += w * hcat[(size_t)s_nbr[i] * NCOLS + c];
    }
    float out = (acc > 0.f) ? acc : (__expf(acc) - 1.f);   // ELU
    xout[(size_t)t * NCOLS + c] = out;
}

extern "C" void kernel_launch(void* const* d_in, const int* in_sizes, int n_in,
                              void* d_out, int out_size, void* d_ws, size_t ws_size,
                              hipStream_t stream) {
    const float* x     = (const float*)d_in[0];
    const float* W0    = (const float*)d_in[1];
    const float* a0    = (const float*)d_in[2];
    const float* W1    = (const float*)d_in[3];
    const float* a1    = (const float*)d_in[4];
    const float* linW  = (const float*)d_in[5];
    const float* linb  = (const float*)d_in[6];
    const int* adj1_nbr = (const int*)d_in[8];
    const int* adj0_nbr = (const int*)d_in[10];

    char* ws = (char*)d_ws;
    size_t off = 0;
    auto alloc = [&](size_t bytes) -> void* {
        void* p = ws + off;
        off = (off + bytes + 255) & ~(size_t)255;
        return p;
    };
    unsigned short* xb  = (unsigned short*)alloc((size_t)N2 * 256 * 2);
    unsigned short* Bp0 = (unsigned short*)alloc(65536 * 2);
    unsigned short* Bp1 = (unsigned short*)alloc(65536 * 2);
    unsigned short* BpL = (unsigned short*)alloc(65536 * 2);
    float* h1  = (float*)alloc((size_t)N2 * 256 * 4);
    float* st1 = (float*)alloc((size_t)NHEAD * N2 * 4);
    float* sn1 = (float*)alloc((size_t)NHEAD * N2 * 4);
    float* x1  = (float*)alloc((size_t)N1 * 256 * 4);
    unsigned short* x1b = (unsigned short*)alloc((size_t)N1 * 256 * 2);
    float* h2  = (float*)alloc((size_t)N1 * 256 * 4);
    float* st2 = (float*)alloc((size_t)NHEAD * N1 * 4);
    float* sn2 = (float*)alloc((size_t)NHEAD * N1 * 4);
    float* x2  = (float*)alloc((size_t)N0 * 256 * 4);
    unsigned short* x2b = (unsigned short*)alloc((size_t)N0 * 256 * 2);

    const dim3 blk(256);

    // ---- layer 1 ----
    cvt_f32_bf16<<<(N2 * 256 + 255) / 256, blk, 0, stream>>>(x, xb, N2 * 256);
    pack_W_headed<<<65536 / 256, blk, 0, stream>>>(W0, Bp0);
    wmma_gemm_bf16<<<dim3((N2 + 63) / 64, NCOLS / 32), blk, 0, stream>>>(
        xb, Bp0, nullptr, h1, N2, 0);
    node_scores<<<(N2 + 255) / 256, blk, 0, stream>>>(h1, a0, st1, sn1, N2);
    aggregate<<<N1, blk, 0, stream>>>(h1, st1, sn1, adj1_nbr, x1, N1, N2);

    // ---- layer 2 ----
    cvt_f32_bf16<<<(N1 * 256 + 255) / 256, blk, 0, stream>>>(x1, x1b, N1 * 256);
    pack_W_headed<<<65536 / 256, blk, 0, stream>>>(W1, Bp1);
    wmma_gemm_bf16<<<dim3((N1 + 63) / 64, NCOLS / 32), blk, 0, stream>>>(
        x1b, Bp1, nullptr, h2, N1, 0);
    node_scores<<<(N1 + 255) / 256, blk, 0, stream>>>(h2, a1, st2, sn2, N1);
    aggregate<<<N0, blk, 0, stream>>>(h2, st2, sn2, adj0_nbr, x2, N0, N1);

    // ---- final linear + tanh ----
    cvt_f32_bf16<<<(N0 * 256 + 255) / 256, blk, 0, stream>>>(x2, x2b, N0 * 256);
    pack_W_linT<<<65536 / 256, blk, 0, stream>>>(linW, BpL);
    wmma_gemm_bf16<<<dim3((N0 + 63) / 64, NCOLS / 32), blk, 0, stream>>>(
        x2b, BpL, linb, (float*)d_out, N0, 1);
}
